// PointCloudEncoder_46119358824584
// MI455X (gfx1250) — matrix operand order repaired and checked by reading, not compiled, not run on hardware
//
#include <hip/hip_runtime.h>
#include <cstddef>
#include <cstdint>

// ---------------------------------------------------------------------------
// CDNA5 (gfx1250) wave32 WMMA types
// ---------------------------------------------------------------------------
typedef __attribute__((ext_vector_type(16))) __bf16 bf16x16;
typedef __attribute__((ext_vector_type(8)))  float  f32x8;

#define EPS_BN 1e-5f

__device__ __forceinline__ f32x8 wmma_bf16(bf16x16 a, bf16x16 b, f32x8 c) {
  // D = A(16x32 bf16) * B(32x16 bf16) + C(16x16 f32)
  return __builtin_amdgcn_wmma_f32_16x16x32_bf16(
      /*neg_a=*/false, a, /*neg_b=*/false, b,
      /*c_mod=*/(short)0, c, /*reuse_a=*/false, /*reuse_b=*/false);
}

// A fragment from row-major bf16 LDS tile [rows][ld], k-tile already applied
// to the base pointer. ISA 16-bit A layout: lanes 0-15 hold K[0..8)&[16..24),
// lanes 16-31 hold K[8..16)&[24..32) of row (lane&15)+rowBase.
__device__ __forceinline__ bf16x16 lds_a_frag(const __bf16* s, int ld,
                                              int rowBase, int lane) {
  const __bf16* p = s + (size_t)(rowBase + (lane & 15)) * ld + ((lane < 16) ? 0 : 8);
  bf16x16 f;
#pragma unroll
  for (int j = 0; j < 8; ++j) f[j] = p[j];
#pragma unroll
  for (int j = 0; j < 8; ++j) f[8 + j] = p[16 + j];
  return f;
}

// B fragment from pre-packed weights: blob layout [kt][nt][lane][16 bf16].
__device__ __forceinline__ bf16x16 packed_b_frag(const __bf16* wp, int nTiles,
                                                 int kt, int nt, int lane) {
  return *(const bf16x16*)(wp + (((size_t)(kt * nTiles + nt) * 32) + lane) * 16);
}

// ---------------------------------------------------------------------------
// Weight fold (BN) + pack into WMMA B-fragment layout
//   W'(col,kk) = W(col,kk) * gamma(col)/sqrt(var+eps), zero-padded to Kpad
//   b'(col)    = (b - mean) * s + beta
// ---------------------------------------------------------------------------
__global__ void pack_w_kernel(const float* __restrict__ W,
                              const float* __restrict__ g,
                              const float* __restrict__ be,
                              const float* __restrict__ mn,
                              const float* __restrict__ va,
                              const float* __restrict__ bias,  // may be null
                              int Cout, int Cin, int Kpad,
                              __bf16* __restrict__ wp,
                              float* __restrict__ bo) {
  int e = blockIdx.x * 256 + threadIdx.x;
  int total = Kpad * Cout;
  if (e >= total) return;
  int nTiles = Cout >> 4;
  int t    = e >> 9;        // 512 elems per 32x16 tile
  int w    = e & 511;
  int lane = w >> 4;
  int j    = w & 15;
  int kt   = t / nTiles;
  int nt   = t - kt * nTiles;
  int col  = (nt << 4) + (lane & 15);
  int kk   = (kt << 5) + ((lane < 16) ? 0 : 8) + (j & 7) + ((j & 8) ? 16 : 0);
  float s  = g[col] * rsqrtf(va[col] + EPS_BN);
  float wv = (kk < Cin) ? W[(size_t)col * Cin + kk] * s : 0.f;
  wp[e] = (__bf16)wv;
  if (e < Cout) {
    float sc = g[e] * rsqrtf(va[e] + EPS_BN);
    float b0 = bias ? bias[e] : 0.f;
    bo[e] = (b0 - mn[e]) * sc + be[e];
  }
}

// ---------------------------------------------------------------------------
// Farthest point sampling: one workgroup (256 threads) per (b,t) instance.
// Writes gathered anchor coordinates (BT,M,3). Matches jnp.argmax (first max).
// ---------------------------------------------------------------------------
template <int NMAX>
__global__ __launch_bounds__(256) void fps_kernel(const float* __restrict__ xyz,
                                                  float* __restrict__ anch,
                                                  int N, int M) {
  int bt = blockIdx.x;
  const float* p = xyz + (size_t)bt * N * 3;
  float* a = anch + (size_t)bt * M * 3;
  __shared__ float sd[NMAX];
  __shared__ float rv[256];
  __shared__ int ri[256];
  __shared__ int s_last;
  int tid = threadIdx.x;
  for (int n = tid; n < N; n += 256) sd[n] = 1e10f;
  if (tid == 0) { s_last = 0; a[0] = p[0]; a[1] = p[1]; a[2] = p[2]; }
  __syncthreads();
  for (int s = 1; s < M; ++s) {
    int last = s_last;
    float lx = p[last * 3 + 0], ly = p[last * 3 + 1], lz = p[last * 3 + 2];
    float bv = -1.f; int bi = 0x7fffffff;
    for (int n = tid; n < N; n += 256) {
      float dx = p[n * 3 + 0] - lx;
      float dy = p[n * 3 + 1] - ly;
      float dz = p[n * 3 + 2] - lz;
      float d  = dx * dx + dy * dy + dz * dz;
      float nd = fminf(sd[n], d);
      sd[n] = nd;
      if (nd > bv) { bv = nd; bi = n; }  // local scan is ascending in n
    }
    rv[tid] = bv; ri[tid] = bi;
    __syncthreads();
    for (int off = 128; off > 0; off >>= 1) {
      if (tid < off) {
        float ov = rv[tid + off]; int oi = ri[tid + off];
        if (ov > rv[tid] || (ov == rv[tid] && oi < ri[tid])) { rv[tid] = ov; ri[tid] = oi; }
      }
      __syncthreads();
    }
    if (tid == 0) {
      int nx = ri[0];
      s_last = nx;
      a[s * 3 + 0] = p[nx * 3 + 0];
      a[s * 3 + 1] = p[nx * 3 + 1];
      a[s * 3 + 2] = p[nx * 3 + 2];
    }
    __syncthreads();
  }
}

// ---------------------------------------------------------------------------
// Ball query: one thread per (bt, dt, m). Collect first K in-radius indices in
// ascending order; pad with first-valid (or 0 if none) — matches reference.
// ---------------------------------------------------------------------------
__global__ void bq_kernel(const float* __restrict__ anch,  // (BT,M,3)
                          const float* __restrict__ pts,   // (B,T,N,3) frames
                          int* __restrict__ idx,           // (BT,NDT,M,32)
                          int BT, int T, int M, int N, float r2, int ndt) {
  int gid = blockIdx.x * 256 + threadIdx.x;
  int total = BT * ndt * M;
  if (gid >= total) return;
  int m   = gid % M;
  int r   = gid / M;
  int dtI = r % ndt;
  int bt  = r / ndt;
  int b = bt / T, t = bt - b * T;
  int srcT = t + dtI - (ndt >> 1);
  srcT = srcT < 0 ? 0 : (srcT >= T ? T - 1 : srcT);
  const float* P = pts + ((size_t)(b * T + srcT)) * N * 3;
  const float* A = anch + ((size_t)bt * M + m) * 3;
  float ax = A[0], ay = A[1], az = A[2];
  int* o = idx + (((size_t)bt * ndt + dtI) * M + m) * 32;
  int cnt = 0, first = -1;
  for (int n = 0; n < N; ++n) {
    float dx = ax - P[n * 3 + 0];
    float dy = ay - P[n * 3 + 1];
    float dz = az - P[n * 3 + 2];
    float d2 = dx * dx + dy * dy + dz * dz;
    if (d2 < r2) {
      if (first < 0) first = n;
      if (cnt < 32) o[cnt++] = n;
    }
  }
  int fill = (first >= 0) ? first : 0;
  while (cnt < 32) o[cnt++] = fill;
}

// ---------------------------------------------------------------------------
// WMMA stage helpers (rows = 32 neighbors -> two 16-row tiles, cols tiled x16)
// ---------------------------------------------------------------------------
__device__ __forceinline__ void stage_mm(const __bf16* s_in, int ldin, int kT,
                                         const __bf16* wp, const float* bias,
                                         int nT, __bf16* s_out, int ldout,
                                         int wave, int lane) {
  for (int nt = wave; nt < nT; nt += 4) {
    f32x8 c0 = {}, c1 = {};
    for (int kt = 0; kt < kT; ++kt) {
      bf16x16 bfr = packed_b_frag(wp, nT, kt, nt, lane);
      bf16x16 a0 = lds_a_frag(s_in + kt * 32, ldin, 0, lane);
      bf16x16 a1 = lds_a_frag(s_in + kt * 32, ldin, 16, lane);
      c0 = wmma_bf16(a0, bfr, c0);
      c1 = wmma_bf16(a1, bfr, c1);
    }
    int col = nt * 16 + (lane & 15);
    float bv = bias[col];
    int ro = (lane < 16) ? 0 : 8;
#pragma unroll
    for (int i = 0; i < 8; ++i) {
      float v0 = fmaxf(c0[i] + bv, 0.f);
      float v1 = fmaxf(c1[i] + bv, 0.f);
      s_out[(size_t)(ro + i) * ldout + col]      = (__bf16)v0;
      s_out[(size_t)(ro + i + 16) * ldout + col] = (__bf16)v1;
    }
  }
}

// h = relu(d*Wd + bd) + relu(a*Wf + bf)
__device__ __forceinline__ void stage_dual(const __bf16* s_d, int ldd, int kTd,
                                           const __bf16* wd, const float* bd,
                                           const __bf16* s_a, int lda, int kTf,
                                           const __bf16* wf, const float* bfb,
                                           int nT, __bf16* s_out, int ldout,
                                           int wave, int lane) {
  for (int nt = wave; nt < nT; nt += 4) {
    f32x8 d0 = {}, d1 = {}, f0 = {}, f1 = {};
    for (int kt = 0; kt < kTd; ++kt) {
      bf16x16 bfr = packed_b_frag(wd, nT, kt, nt, lane);
      d0 = wmma_bf16(lds_a_frag(s_d + kt * 32, ldd, 0, lane), bfr, d0);
      d1 = wmma_bf16(lds_a_frag(s_d + kt * 32, ldd, 16, lane), bfr, d1);
    }
    for (int kt = 0; kt < kTf; ++kt) {
      bf16x16 bfr = packed_b_frag(wf, nT, kt, nt, lane);
      f0 = wmma_bf16(lds_a_frag(s_a + kt * 32, lda, 0, lane), bfr, f0);
      f1 = wmma_bf16(lds_a_frag(s_a + kt * 32, lda, 16, lane), bfr, f1);
    }
    int col = nt * 16 + (lane & 15);
    float bvd = bd[col], bvf = bfb[col];
    int ro = (lane < 16) ? 0 : 8;
#pragma unroll
    for (int i = 0; i < 8; ++i) {
      float v0 = fmaxf(d0[i] + bvd, 0.f) + fmaxf(f0[i] + bvf, 0.f);
      float v1 = fmaxf(d1[i] + bvd, 0.f) + fmaxf(f1[i] + bvf, 0.f);
      s_out[(size_t)(ro + i) * ldout + col]      = (__bf16)v0;
      s_out[(size_t)(ro + i + 16) * ldout + col] = (__bf16)v1;
    }
  }
}

// last MLP layer: relu, max over the 32 neighbor rows, accumulate over dt.
__device__ __forceinline__ void stage_final(const __bf16* s_in, int ldin, int kT,
                                            const __bf16* wp, const float* bias,
                                            int nT, float* s_acc,
                                            int wave, int lane) {
  for (int nt = wave; nt < nT; nt += 4) {
    f32x8 c0 = {}, c1 = {};
    for (int kt = 0; kt < kT; ++kt) {
      bf16x16 bfr = packed_b_frag(wp, nT, kt, nt, lane);
      c0 = wmma_bf16(lds_a_frag(s_in + kt * 32, ldin, 0, lane), bfr, c0);
      c1 = wmma_bf16(lds_a_frag(s_in + kt * 32, ldin, 16, lane), bfr, c1);
    }
    int col = nt * 16 + (lane & 15);
    float bv = bias[col];
    float mx = 0.f;  // relu floor
#pragma unroll
    for (int i = 0; i < 8; ++i) {
      mx = fmaxf(mx, fmaxf(c0[i] + bv, 0.f));
      mx = fmaxf(mx, fmaxf(c1[i] + bv, 0.f));
    }
    mx = fmaxf(mx, __shfl_xor(mx, 16, 32));
    if (lane < 16) s_acc[col] += mx;
  }
}

// ---------------------------------------------------------------------------
// Fused P4D layer: one workgroup (128 threads / 4 wave32s) per (bt, anchor).
// Gathers k=32 neighbors, runs conv_d+conv_f -> mlp0 -> mlp1 (max over k),
// summed over NDT temporal shifts.
// ---------------------------------------------------------------------------
template <int CF_IN, int CF_PAD, int C1, int C2, int C3, int NDT>
__global__ __launch_bounds__(128) void p4d_layer_kernel(
    const float* __restrict__ anch,    // (BT,M,3)
    const float* __restrict__ nbxyz,   // (B,T,N,3)
    const float* __restrict__ nbfeat,  // frames of N*CF_IN, strides fsn/fsc
    const int* __restrict__ idx,       // (BT,NDT,M,32)
    const __bf16* __restrict__ wd, const float* __restrict__ bd,
    const __bf16* __restrict__ wf, const float* __restrict__ bfb,
    const __bf16* __restrict__ w1, const float* __restrict__ b1,
    const __bf16* __restrict__ w2, const float* __restrict__ b2,
    float* __restrict__ fout,          // (BT,M,C3)
    int T, int M, int N, int fsn, int fsc) {
  int bm = blockIdx.x;
  int bt = bm / M, m = bm - bt * M;
  int b = bt / T, t = bt - b * T;
  int tid = threadIdx.x, wave = tid >> 5, lane = tid & 31;

  __shared__ __align__(16) __bf16 s_d[32 * 32];        // disp (K padded 4->32)
  __shared__ __align__(16) __bf16 s_a[32 * CF_PAD];    // gathered features
  __shared__ __align__(16) __bf16 s_h1[32 * C1];
  __shared__ __align__(16) __bf16 s_h2[32 * C2];
  __shared__ float s_acc[C3];
  __shared__ int s_idx[32];
  __shared__ float s_anchor[3];

  for (int i = tid; i < 32 * 32; i += 128) s_d[i] = (__bf16)0.f;
  for (int i = tid; i < 32 * CF_PAD; i += 128) s_a[i] = (__bf16)0.f;
  for (int i = tid; i < C3; i += 128) s_acc[i] = 0.f;
  if (tid < 3) s_anchor[tid] = anch[((size_t)bt * M + m) * 3 + tid];
  __syncthreads();

  for (int dtI = 0; dtI < NDT; ++dtI) {
    int dt = dtI - (NDT >> 1);
    int srcT = t + dt;
    srcT = srcT < 0 ? 0 : (srcT >= T ? T - 1 : srcT);
    size_t frame = (size_t)b * T + srcT;
    const int* ii = idx + (((size_t)bt * NDT + dtI) * M + m) * 32;
    if (tid < 32) s_idx[tid] = ii[tid];
    __syncthreads();
    if (tid < 32) {
      int n = s_idx[tid];
      const float* p = nbxyz + (frame * N + n) * 3;
      s_d[tid * 32 + 0] = (__bf16)(p[0] - s_anchor[0]);
      s_d[tid * 32 + 1] = (__bf16)(p[1] - s_anchor[1]);
      s_d[tid * 32 + 2] = (__bf16)(p[2] - s_anchor[2]);
      s_d[tid * 32 + 3] = (__bf16)((float)dt);
    }
    const float* fbase = nbfeat + frame * (size_t)N * CF_IN;
    for (int e = tid; e < 32 * CF_IN; e += 128) {
      int j = e / CF_IN, c = e - j * CF_IN;
      int n = s_idx[j];
      s_a[j * CF_PAD + c] = (__bf16)fbase[(size_t)n * fsn + (size_t)c * fsc];
    }
    __syncthreads();

    stage_dual(s_d, 32, 1, wd, bd, s_a, CF_PAD, CF_PAD / 32, wf, bfb,
               C1 / 16, s_h1, C1, wave, lane);
    __syncthreads();
    stage_mm(s_h1, C1, C1 / 32, w1, b1, C2 / 16, s_h2, C2, wave, lane);
    __syncthreads();
    stage_final(s_h2, C2, C2 / 32, w2, b2, C3 / 16, s_acc, wave, lane);
    __syncthreads();
  }

  float* o = fout + ((size_t)bt * M + m) * C3;
  for (int c = tid; c < C3; c += 128) o[c] = s_acc[c];
}

// ---------------------------------------------------------------------------
// Final: pos = x2 * Wp' + bp' (no relu), emb = pos + f2, max over M=32 anchors
// One workgroup per (b,t). Output (BT,1024) f32.
// ---------------------------------------------------------------------------
__global__ __launch_bounds__(128) void pos_max_kernel(
    const float* __restrict__ x2,      // (BT,32,3)
    const float* __restrict__ f2,      // (BT,32,1024)
    const __bf16* __restrict__ wp, const float* __restrict__ bp,
    float* __restrict__ out) {         // (BT,1024)
  int bt = blockIdx.x;
  int tid = threadIdx.x, wave = tid >> 5, lane = tid & 31;
  __shared__ __align__(16) __bf16 s_x[32 * 32];
  for (int i = tid; i < 32 * 32; i += 128) s_x[i] = (__bf16)0.f;
  __syncthreads();
  if (tid < 32) {
    const float* p = x2 + ((size_t)bt * 32 + tid) * 3;
    s_x[tid * 32 + 0] = (__bf16)p[0];
    s_x[tid * 32 + 1] = (__bf16)p[1];
    s_x[tid * 32 + 2] = (__bf16)p[2];
  }
  __syncthreads();
  const float* f2b = f2 + (size_t)bt * 32 * 1024;
  for (int nt = wave; nt < 64; nt += 4) {
    f32x8 c0 = {}, c1 = {};
    bf16x16 bfr = packed_b_frag(wp, 64, 0, nt, lane);
    c0 = wmma_bf16(lds_a_frag(s_x, 32, 0, lane), bfr, c0);
    c1 = wmma_bf16(lds_a_frag(s_x, 32, 16, lane), bfr, c1);
    int col = nt * 16 + (lane & 15);
    float bv = bp[col];
    int ro = (lane < 16) ? 0 : 8;
    float mx = -3.4e38f;
#pragma unroll
    for (int i = 0; i < 8; ++i) {
      mx = fmaxf(mx, c0[i] + bv + f2b[(size_t)(ro + i) * 1024 + col]);
      mx = fmaxf(mx, c1[i] + bv + f2b[(size_t)(ro + i + 16) * 1024 + col]);
    }
    mx = fmaxf(mx, __shfl_xor(mx, 16, 32));
    if (lane < 16) out[(size_t)bt * 1024 + col] = mx;
  }
}

// ---------------------------------------------------------------------------
// Host launcher
// ---------------------------------------------------------------------------
extern "C" void kernel_launch(void* const* d_in, const int* in_sizes, int n_in,
                              void* d_out, int out_size, void* d_ws, size_t ws_size,
                              hipStream_t stream) {
  (void)in_sizes; (void)n_in; (void)out_size; (void)ws_size;
  constexpr int B = 4, T = 8, N = 1024, BT = B * T;
  constexpr int M1 = 128, M2 = 32;

  const float* xyzs     = (const float*)d_in[0];  // (B,T,N,3)
  const float* features = (const float*)d_in[1];  // (B,T,2,N)

  // Flattened param order: W [,b], gamma, beta, mean, var per conv layer.
  // p1: conv_d(2..6) conv_f(7..11) mlp0(12..16) mlp1(17..21)
  // p2: conv_d(22..26) conv_f(27..31) mlp0(32..36) mlp1(37..41)
  // ppos: 42..47 (has bias)
  struct LayerDesc { int base; bool bias; int Cout, Cin, Kpad; };
  const LayerDesc L[9] = {
      {2,  false, 32,   4,  32},   // p1.conv_d
      {7,  false, 32,   2,  32},   // p1.conv_f
      {12, false, 64,   32, 32},   // p1.mlp0
      {17, false, 128,  64, 64},   // p1.mlp1
      {22, false, 256,  4,  32},   // p2.conv_d
      {27, false, 256,  128,128},  // p2.conv_f
      {32, false, 512,  256,256},  // p2.mlp0
      {37, false, 1024, 512,512},  // p2.mlp1
      {42, true,  1024, 3,  32},   // ppos
  };

  // --- workspace bump allocator (256B aligned) ---
  char* ws = (char*)d_ws;
  size_t off = 0;
  auto alloc = [&](size_t bytes) -> char* {
    char* p = ws + off;
    off += (bytes + 255) & ~(size_t)255;
    return p;
  };

  __bf16* wpk[9];
  float*  bof[9];
  for (int i = 0; i < 9; ++i) {
    wpk[i] = (__bf16*)alloc((size_t)L[i].Kpad * L[i].Cout * sizeof(__bf16));
    bof[i] = (float*)alloc((size_t)L[i].Cout * sizeof(float));
  }
  float* x1  = (float*)alloc((size_t)BT * M1 * 3 * sizeof(float));
  float* f1  = (float*)alloc((size_t)BT * M1 * 128 * sizeof(float));
  float* x2  = (float*)alloc((size_t)BT * M2 * 3 * sizeof(float));
  float* f2  = (float*)alloc((size_t)BT * M2 * 1024 * sizeof(float));
  int*   id1 = (int*)alloc((size_t)BT * 1 * M1 * 32 * sizeof(int));
  int*   id2 = (int*)alloc((size_t)BT * 3 * M2 * 32 * sizeof(int));

  // --- fold + pack all weights ---
  for (int i = 0; i < 9; ++i) {
    const LayerDesc& d = L[i];
    int o = d.bias ? 1 : 0;
    const float* W    = (const float*)d_in[d.base];
    const float* bias = d.bias ? (const float*)d_in[d.base + 1] : nullptr;
    const float* g    = (const float*)d_in[d.base + 1 + o];
    const float* be   = (const float*)d_in[d.base + 2 + o];
    const float* mn   = (const float*)d_in[d.base + 3 + o];
    const float* va   = (const float*)d_in[d.base + 4 + o];
    int total = d.Kpad * d.Cout;
    pack_w_kernel<<<(total + 255) / 256, 256, 0, stream>>>(
        W, g, be, mn, va, bias, d.Cout, d.Cin, d.Kpad, wpk[i], bof[i]);
  }

  // --- layer 1: FPS -> ball query (r=0.9) -> fused conv/MLP ---
  fps_kernel<1024><<<BT, 256, 0, stream>>>(xyzs, x1, N, M1);
  {
    int total = BT * 1 * M1;
    bq_kernel<<<(total + 255) / 256, 256, 0, stream>>>(
        x1, xyzs, id1, BT, T, M1, N, 0.9f * 0.9f, 1);
  }
  // features (B,T,2,N): fsn=1, fsc=N
  p4d_layer_kernel<2, 32, 32, 64, 128, 1><<<BT * M1, 128, 0, stream>>>(
      x1, xyzs, features, id1,
      wpk[0], bof[0], wpk[1], bof[1], wpk[2], bof[2], wpk[3], bof[3],
      f1, T, M1, N, /*fsn=*/1, /*fsc=*/N);

  // --- layer 2: FPS on x1 -> ball query (r=1.8, 3 dts) -> fused conv/MLP ---
  fps_kernel<128><<<BT, 256, 0, stream>>>(x1, x2, M1, M2);
  {
    int total = BT * 3 * M2;
    bq_kernel<<<(total + 255) / 256, 256, 0, stream>>>(
        x2, x1, id2, BT, T, M2, M1, 1.8f * 1.8f, 3);
  }
  // f1 (BT, M1, 128): fsn=128, fsc=1
  p4d_layer_kernel<128, 128, 256, 512, 1024, 3><<<BT * M2, 128, 0, stream>>>(
      x2, x1, f1, id2,
      wpk[4], bof[4], wpk[5], bof[5], wpk[6], bof[6], wpk[7], bof[7],
      f2, T, M2, M1, /*fsn=*/128, /*fsc=*/1);

  // --- final: pos conv (no relu) + f2, max over anchors ---
  pos_max_kernel<<<BT, 128, 0, stream>>>(x2, f2, wpk[8], bof[8], (float*)d_out);
}
